// SoftMoELayer_10471130267897
// MI455X (gfx1250) — matrix operand order repaired and checked by reading, not compiled, not run on hardware
//
#include <hip/hip_runtime.h>
#include <hip/hip_bf16.h>

// ---------------------------------------------------------------------------
// Soft-MoE forward for MI455X (gfx1250): WMMA bf16 GEMM pipeline, v3.
// B=8, T=4096, D=1024, N_EXPERTS=64, N_SLOTS=8, HID=256.
// v3: dedicated bf16 token buffer (no f32 A-path in router), launch_bounds
//     to lift the VGPR cap and eliminate inner-loop spills observed in v2.
// ---------------------------------------------------------------------------

typedef __bf16 bf16_t;
typedef bf16_t v16bf __attribute__((ext_vector_type(16)));
typedef float  v8f   __attribute__((ext_vector_type(8)));

#define BT_      32768
#define D_       1024
#define SLOTS_   512
#define NEXP_    64
#define HID_     256
#define KSPLIT_  4
#define KCHUNK_  (BT_ / KSPLIT_)   // 8192

__device__ __forceinline__ int lane_id() { return threadIdx.x & 31; }

union FragU { uint4 q[2]; v16bf v; };

// ---- A fragment: 16x32 bf16 tile from row-major [M x K] memory (K contig).
// Lane l<16: row M=l, K={kb..kb+7, kb+16..kb+23} with kb=0; lanes 16-31: kb=8.
__device__ __forceinline__ v16bf load_frag_row16_k32(const bf16_t* base, int ld) {
  int l   = lane_id();
  int row = l & 15;
  int kb  = (l >> 4) << 3;          // 0 or 8
  const bf16_t* p = base + (size_t)row * ld + kb;
  FragU f;
  f.q[0] = *reinterpret_cast<const uint4*>(p);
  f.q[1] = *reinterpret_cast<const uint4*>(p + 16);
  return f.v;
}

// Same but duplicate rows 8..15 onto 0..7 (M-padding for the 8-slot experts).
__device__ __forceinline__ v16bf load_frag_row16_k32_pad8(const bf16_t* base, int ld) {
  int l   = lane_id();
  int row = (l & 15) & 7;
  int kb  = (l >> 4) << 3;
  const bf16_t* p = base + (size_t)row * ld + kb;
  FragU f;
  f.q[0] = *reinterpret_cast<const uint4*>(p);
  f.q[1] = *reinterpret_cast<const uint4*>(p + 16);
  return f.v;
}

// ---- B fragment: 32x16 (KxN) tile taken from N-transposed storage [N x K]
// (K contiguous). Lane l<16: col N=l, K={kb..kb+7, kb+16..kb+23} with kb=0; lanes 16-31: kb=8.
__device__ __forceinline__ v16bf load_frag_col16_k32(const bf16_t* baseNT, int ld) {
  int l  = lane_id();
  int n  = l & 15;
  int kb = (l >> 4) << 3;           // 0 or 8
  const bf16_t* p = baseNT + (size_t)n * ld + kb;
  FragU f;
  f.q[0] = *reinterpret_cast<const uint4*>(p);
  f.q[1] = *reinterpret_cast<const uint4*>(p + 16);
  return f.v;
}

__device__ __forceinline__ v8f wmma_bf16(v16bf a, v16bf b, v8f c) {
  return __builtin_amdgcn_wmma_f32_16x16x32_bf16(false, a, false, b, (short)0, c,
                                                 false, false);
}

// ---- C/D 16x16 f32 tile store: VGPR r -> M = r + 8*(lane>=16), N = lane&15.
__device__ __forceinline__ void store_tile_f32(float* base, int ld, v8f c) {
  int l  = lane_id();
  int n  = l & 15;
  int mo = (l >> 4) << 3;
  float* p = base + (size_t)mo * ld + n;
#pragma unroll
  for (int r = 0; r < 8; ++r) p[(size_t)r * ld] = c[r];
}

// ===========================================================================
// 0) Plain fp32 -> bf16 convert (8 elements/thread, b128 in / b128 out).
// ===========================================================================
__global__ void k_cvt_bf16(const float* __restrict__ in,
                           bf16_t* __restrict__ out) {
  size_t i = ((size_t)blockIdx.x * 256 + threadIdx.x) * 8;
  float4 a = *reinterpret_cast<const float4*>(in + i);
  float4 b = *reinterpret_cast<const float4*>(in + i + 4);
  union { uint4 q; bf16_t h[8]; } u;
  u.h[0] = (bf16_t)a.x; u.h[1] = (bf16_t)a.y;
  u.h[2] = (bf16_t)a.z; u.h[3] = (bf16_t)a.w;
  u.h[4] = (bf16_t)b.x; u.h[5] = (bf16_t)b.y;
  u.h[6] = (bf16_t)b.z; u.h[7] = (bf16_t)b.w;
  *reinterpret_cast<uint4*>(out + i) = u.q;
}

// ===========================================================================
// 1) Tiled transpose + fp32->bf16 convert: in [R x C] f32 -> out [C x R] bf16.
// ===========================================================================
__global__ void k_transpose_cvt(const float* __restrict__ in,
                                bf16_t* __restrict__ out, int R, int C) {
  __shared__ float tile[32][33];
  size_t off = (size_t)blockIdx.z * R * C;
  in  += off;
  out += off;
  int cb = blockIdx.x * 32;
  int rb = blockIdx.y * 32;
  int tx = threadIdx.x & 31;
  int tg = threadIdx.x >> 5;       // 0..7, four rows each
#pragma unroll
  for (int i = 0; i < 4; ++i) {
    int r = rb + tg * 4 + i;
    tile[tg * 4 + i][tx] = in[(size_t)r * C + cb + tx];
  }
  __syncthreads();
#pragma unroll
  for (int i = 0; i < 4; ++i) {
    int ro = tg * 4 + i;           // input-col index inside tile
    out[(size_t)(cb + ro) * R + rb + tx] = (bf16_t)tile[tx][ro];
  }
}

// ===========================================================================
// 2) Router GEMM: mat[BT x S] = tokens_bf @ em.  Wave = 2M x 4N tiles,
//    software-pipelined K loop. launch_bounds(256,1) lifts the VGPR cap so
//    the double-buffered fragment set stays in registers (no spills).
// ===========================================================================
__global__ void __launch_bounds__(256, 1)
k_router_gemm(const bf16_t* __restrict__ tokens_bf,
              const bf16_t* __restrict__ em_t,
              float* __restrict__ mat) {
  int w     = threadIdx.x >> 5;
  int mbase = (blockIdx.y * 8 + w) * 32;
  int nbase = blockIdx.x * 64;
  const bf16_t* pa0 = tokens_bf + (size_t)mbase * D_;
  const bf16_t* pa1 = pa0 + (size_t)16 * D_;
  const bf16_t* p0  = em_t + (size_t)(nbase +  0) * D_;
  const bf16_t* p1  = em_t + (size_t)(nbase + 16) * D_;
  const bf16_t* p2  = em_t + (size_t)(nbase + 32) * D_;
  const bf16_t* p3  = em_t + (size_t)(nbase + 48) * D_;

  v8f acc[8] = {};
  v16bf a0 = load_frag_row16_k32(pa0, D_);
  v16bf a1 = load_frag_row16_k32(pa1, D_);
  v16bf b0 = load_frag_col16_k32(p0, D_);
  v16bf b1 = load_frag_col16_k32(p1, D_);
  v16bf b2 = load_frag_col16_k32(p2, D_);
  v16bf b3 = load_frag_col16_k32(p3, D_);

  for (int k = 0; k < D_ - 32; k += 32) {
    int kn = k + 32;
    __builtin_prefetch(pa0 + kn + 32, 0, 3);
    v16bf a0n = load_frag_row16_k32(pa0 + kn, D_);
    v16bf a1n = load_frag_row16_k32(pa1 + kn, D_);
    v16bf b0n = load_frag_col16_k32(p0 + kn, D_);
    v16bf b1n = load_frag_col16_k32(p1 + kn, D_);
    v16bf b2n = load_frag_col16_k32(p2 + kn, D_);
    v16bf b3n = load_frag_col16_k32(p3 + kn, D_);
    acc[0] = wmma_bf16(a0, b0, acc[0]);
    acc[1] = wmma_bf16(a0, b1, acc[1]);
    acc[2] = wmma_bf16(a0, b2, acc[2]);
    acc[3] = wmma_bf16(a0, b3, acc[3]);
    acc[4] = wmma_bf16(a1, b0, acc[4]);
    acc[5] = wmma_bf16(a1, b1, acc[5]);
    acc[6] = wmma_bf16(a1, b2, acc[6]);
    acc[7] = wmma_bf16(a1, b3, acc[7]);
    a0 = a0n; a1 = a1n; b0 = b0n; b1 = b1n; b2 = b2n; b3 = b3n;
  }
  acc[0] = wmma_bf16(a0, b0, acc[0]);
  acc[1] = wmma_bf16(a0, b1, acc[1]);
  acc[2] = wmma_bf16(a0, b2, acc[2]);
  acc[3] = wmma_bf16(a0, b3, acc[3]);
  acc[4] = wmma_bf16(a1, b0, acc[4]);
  acc[5] = wmma_bf16(a1, b1, acc[5]);
  acc[6] = wmma_bf16(a1, b2, acc[6]);
  acc[7] = wmma_bf16(a1, b3, acc[7]);

  float* po  = mat + (size_t)mbase * SLOTS_ + nbase;
  float* po2 = po + (size_t)16 * SLOTS_;
  store_tile_f32(po  +  0, SLOTS_, acc[0]);
  store_tile_f32(po  + 16, SLOTS_, acc[1]);
  store_tile_f32(po  + 32, SLOTS_, acc[2]);
  store_tile_f32(po  + 48, SLOTS_, acc[3]);
  store_tile_f32(po2 +  0, SLOTS_, acc[4]);
  store_tile_f32(po2 + 16, SLOTS_, acc[5]);
  store_tile_f32(po2 + 32, SLOTS_, acc[6]);
  store_tile_f32(po2 + 48, SLOTS_, acc[7]);
}

// ===========================================================================
// 3) Row softmax stats (axis=1, 512 wide): one wave per token row.
// ===========================================================================
__global__ void k_row_stats(const float* __restrict__ mat,
                            float* __restrict__ rowmax,
                            float* __restrict__ rowinv) {
  int w = threadIdx.x >> 5, l = threadIdx.x & 31;
  int t = blockIdx.x * 8 + w;
  const float* row = mat + (size_t)t * SLOTS_;
  float x[16], m = -3.4e38f;
#pragma unroll
  for (int i = 0; i < 16; ++i) { x[i] = row[l + 32 * i]; m = fmaxf(m, x[i]); }
#pragma unroll
  for (int o = 16; o > 0; o >>= 1) m = fmaxf(m, __shfl_xor(m, o, 32));
  float s = 0.f;
#pragma unroll
  for (int i = 0; i < 16; ++i) s += __expf(x[i] - m);
#pragma unroll
  for (int o = 16; o > 0; o >>= 1) s += __shfl_xor(s, o, 32);
  if (l == 0) { rowmax[t] = m; rowinv[t] = 1.f / s; }
}

// ===========================================================================
// 4a) Column softmax partials: block = 32 consecutive columns x 1024 tokens.
// ===========================================================================
__global__ void k_col_partial(const float* __restrict__ mat,
                              float* __restrict__ pmax,
                              float* __restrict__ psum) {
  int sb = blockIdx.x * 32;
  int ch = blockIdx.y;                       // token chunk, 1024 tokens each
  int w  = threadIdx.x >> 5, l = threadIdx.x & 31;
  int s  = sb + l;
  int t0 = ch * 1024;
  float m = -3.4e38f, sum = 0.f;
  for (int t = t0 + w; t < t0 + 1024; t += 8) {
    float x  = mat[(size_t)t * SLOTS_ + s];
    float mn = fmaxf(m, x);
    sum = sum * __expf(m - mn) + __expf(x - mn);
    m = mn;
  }
  __shared__ float sm[8][32], ss[8][32];
  sm[w][l] = m; ss[w][l] = sum;
  __syncthreads();
  if (w == 0) {
    float M = m, S = sum;
#pragma unroll
    for (int i = 1; i < 8; ++i) {
      float mn = fmaxf(M, sm[i][l]);
      S = S * __expf(M - mn) + ss[i][l] * __expf(sm[i][l] - mn);
      M = mn;
    }
    pmax[(size_t)ch * SLOTS_ + s] = M;
    psum[(size_t)ch * SLOTS_ + s] = S;
  }
}

// 4b) finalize: combine 32 chunk partials per column.
__global__ void k_col_final(const float* __restrict__ pmax,
                            const float* __restrict__ psum,
                            float* __restrict__ colmax,
                            float* __restrict__ colinv) {
  int s = blockIdx.x * 256 + threadIdx.x;
  float M = pmax[s], S = psum[s];
  for (int i = 1; i < BT_ / 1024; ++i) {
    float m2 = pmax[(size_t)i * SLOTS_ + s];
    float s2 = psum[(size_t)i * SLOTS_ + s];
    float mn = fmaxf(M, m2);
    S = S * __expf(M - mn) + s2 * __expf(m2 - mn);
    M = mn;
  }
  colmax[s] = M; colinv[s] = 1.f / S;
}

// ===========================================================================
// 5) Probability pass with LDS 32x32 transpose (coalesced row_probs^T).
// ===========================================================================
__global__ void k_probs(const float* __restrict__ mat,
                        const float* __restrict__ rowmax,
                        const float* __restrict__ rowinv,
                        const float* __restrict__ colmax,
                        const float* __restrict__ colinv,
                        bf16_t* __restrict__ colprobs,
                        bf16_t* __restrict__ rowprobsT) {
  __shared__ bf16_t tile[32][34];
  int tb = blockIdx.x * 32;              // token base
  int sb = blockIdx.y * 32;              // slot base
  int tx = threadIdx.x & 31;
  int tg = threadIdx.x >> 5;             // 0..7
  float cm = colmax[sb + tx], ci = colinv[sb + tx];
#pragma unroll
  for (int i = 0; i < 4; ++i) {
    int tl = tg * 4 + i;
    int t  = tb + tl;
    float x  = mat[(size_t)t * SLOTS_ + sb + tx];
    float rm = rowmax[t], ri = rowinv[t];
    colprobs[(size_t)t * SLOTS_ + sb + tx] = (bf16_t)(__expf(x - rm) * ri);
    tile[tl][tx] = (bf16_t)(__expf(x - cm) * ci);
  }
  __syncthreads();
#pragma unroll
  for (int i = 0; i < 4; ++i) {
    int sl = tg * 4 + i;
    rowprobsT[(size_t)(sb + sl) * BT_ + tb + tx] = tile[tx][sl];
  }
}

// ===========================================================================
// 6a) Dispatch GEMM, K-split x4: partial x_hat chunks in f32 (deterministic).
//     Wave = 1M x 4N tiles, software-pipelined.
// ===========================================================================
__global__ void __launch_bounds__(256, 1)
k_dispatch_gemm(const bf16_t* __restrict__ rowprobsT,
                const bf16_t* __restrict__ tokens_t,
                float* __restrict__ xpart) {
  int w     = threadIdx.x >> 5;
  int mbase = (blockIdx.y * 8 + w) * 16;
  int nbase = blockIdx.x * 64;
  int kbase = blockIdx.z * KCHUNK_;
  const bf16_t* pa = rowprobsT + (size_t)mbase * BT_ + kbase;
  const bf16_t* p0 = tokens_t + (size_t)(nbase +  0) * BT_ + kbase;
  const bf16_t* p1 = tokens_t + (size_t)(nbase + 16) * BT_ + kbase;
  const bf16_t* p2 = tokens_t + (size_t)(nbase + 32) * BT_ + kbase;
  const bf16_t* p3 = tokens_t + (size_t)(nbase + 48) * BT_ + kbase;

  v8f acc[4] = {};
  v16bf a  = load_frag_row16_k32(pa, BT_);
  v16bf b0 = load_frag_col16_k32(p0, BT_);
  v16bf b1 = load_frag_col16_k32(p1, BT_);
  v16bf b2 = load_frag_col16_k32(p2, BT_);
  v16bf b3 = load_frag_col16_k32(p3, BT_);

  for (int k = 0; k < KCHUNK_ - 32; k += 32) {
    int kn = k + 32;
    __builtin_prefetch(pa + kn + 96, 0, 3);
    __builtin_prefetch(p0 + kn + 96, 0, 3);
    v16bf an  = load_frag_row16_k32(pa + kn, BT_);
    v16bf b0n = load_frag_col16_k32(p0 + kn, BT_);
    v16bf b1n = load_frag_col16_k32(p1 + kn, BT_);
    v16bf b2n = load_frag_col16_k32(p2 + kn, BT_);
    v16bf b3n = load_frag_col16_k32(p3 + kn, BT_);
    acc[0] = wmma_bf16(a, b0, acc[0]);
    acc[1] = wmma_bf16(a, b1, acc[1]);
    acc[2] = wmma_bf16(a, b2, acc[2]);
    acc[3] = wmma_bf16(a, b3, acc[3]);
    a = an; b0 = b0n; b1 = b1n; b2 = b2n; b3 = b3n;
  }
  acc[0] = wmma_bf16(a, b0, acc[0]);
  acc[1] = wmma_bf16(a, b1, acc[1]);
  acc[2] = wmma_bf16(a, b2, acc[2]);
  acc[3] = wmma_bf16(a, b3, acc[3]);

  float* po = xpart + (size_t)blockIdx.z * SLOTS_ * D_ + (size_t)mbase * D_ + nbase;
  store_tile_f32(po +  0, D_, acc[0]);
  store_tile_f32(po + 16, D_, acc[1]);
  store_tile_f32(po + 32, D_, acc[2]);
  store_tile_f32(po + 48, D_, acc[3]);
}

// 6b) Reduce K-split partials, convert to bf16 x_hat [S x D].
__global__ void k_xhat_reduce(const float* __restrict__ xpart,
                              bf16_t* __restrict__ xhat) {
  int i = blockIdx.x * 256 + threadIdx.x;
  float v = xpart[i];
#pragma unroll
  for (int c = 1; c < KSPLIT_; ++c) v += xpart[(size_t)c * SLOTS_ * D_ + i];
  xhat[i] = (bf16_t)v;
}

// ===========================================================================
// 7) Per-expert MLP, one workgroup (8 waves) per expert. M-tile = 16 (8 real
//    slots + 8 duplicated pad rows, pad results discarded). Hidden activations
//    staged in LDS as bf16. y_hat written transposed [D x SLOTS] with +b3.
// ===========================================================================
__global__ void __launch_bounds__(256, 1)
k_expert_mlp(const bf16_t* __restrict__ xhat,
             const bf16_t* __restrict__ w1t, const float* __restrict__ b1,
             const bf16_t* __restrict__ w2t, const float* __restrict__ b2,
             const bf16_t* __restrict__ w3t, const float* __restrict__ b3,
             bf16_t* __restrict__ yhat_t) {
  __shared__ __align__(16) bf16_t h1[16 * HID_];
  __shared__ __align__(16) bf16_t h2[16 * HID_];
  int e = blockIdx.x;
  int w = threadIdx.x >> 5;
  int l = threadIdx.x & 31;
  int n  = l & 15;
  int mo = (l >> 4) << 3;

  // ---- stage 1: h1 = relu(x[16 x 1024] @ W1 + b1), N=256 -> 2 tiles/wave
  {
    v8f c0 = {}, c1 = {};
    const bf16_t* pa = xhat + (size_t)(e * 8) * D_;
    const bf16_t* p0 = w1t + (size_t)e * HID_ * D_ + (size_t)(w * 32) * D_;
    const bf16_t* p1 = p0 + (size_t)16 * D_;
    for (int k = 0; k < D_; k += 32) {
      v16bf a   = load_frag_row16_k32_pad8(pa + k, D_);
      v16bf b0  = load_frag_col16_k32(p0 + k, D_);
      v16bf b1f = load_frag_col16_k32(p1 + k, D_);
      c0 = wmma_bf16(a, b0, c0);
      c1 = wmma_bf16(a, b1f, c1);
    }
    float bia0 = b1[e * HID_ + w * 32 + n];
    float bia1 = b1[e * HID_ + w * 32 + 16 + n];
#pragma unroll
    for (int r = 0; r < 8; ++r) {
      h1[(mo + r) * HID_ + w * 32 + n]      = (bf16_t)fmaxf(c0[r] + bia0, 0.f);
      h1[(mo + r) * HID_ + w * 32 + 16 + n] = (bf16_t)fmaxf(c1[r] + bia1, 0.f);
    }
  }
  __syncthreads();

  // ---- stage 2: h2 = relu(h1[16 x 256] @ W2 + b2)
  {
    v8f c0 = {}, c1 = {};
    const bf16_t* p0 = w2t + (size_t)e * HID_ * HID_ + (size_t)(w * 32) * HID_;
    const bf16_t* p1 = p0 + (size_t)16 * HID_;
    for (int k = 0; k < HID_; k += 32) {
      v16bf a   = load_frag_row16_k32(h1 + k, HID_);
      v16bf b0  = load_frag_col16_k32(p0 + k, HID_);
      v16bf b1f = load_frag_col16_k32(p1 + k, HID_);
      c0 = wmma_bf16(a, b0, c0);
      c1 = wmma_bf16(a, b1f, c1);
    }
    float bia0 = b2[e * HID_ + w * 32 + n];
    float bia1 = b2[e * HID_ + w * 32 + 16 + n];
#pragma unroll
    for (int r = 0; r < 8; ++r) {
      h2[(mo + r) * HID_ + w * 32 + n]      = (bf16_t)fmaxf(c0[r] + bia0, 0.f);
      h2[(mo + r) * HID_ + w * 32 + 16 + n] = (bf16_t)fmaxf(c1[r] + bia1, 0.f);
    }
  }
  __syncthreads();

  // ---- stage 3: y_hat = h2[16 x 256] @ W3 + b3, N=1024 -> 8 tiles/wave.
  for (int j = 0; j < 8; ++j) {
    int n0 = w * 128 + j * 16;
    v8f c = {};
    const bf16_t* pb = w3t + (size_t)e * D_ * HID_ + (size_t)n0 * HID_;
    for (int k = 0; k < HID_; k += 32) {
      v16bf a = load_frag_row16_k32(h2 + k, HID_);
      v16bf b = load_frag_col16_k32(pb + k, HID_);
      c = wmma_bf16(a, b, c);
    }
    if (mo == 0) {                       // rows 0..7 = the 8 valid slots
      float bias = b3[e * D_ + n0 + n];
#pragma unroll
      for (int r = 0; r < 8; ++r) {
        yhat_t[(size_t)(n0 + n) * SLOTS_ + e * 8 + r] = (bf16_t)(c[r] + bias);
      }
    }
  }
}

// ===========================================================================
// 8) Combine GEMM: y[BT x D] = col_probs[BT x S] @ y_hat (yhat_t is [D x S]).
//    Wave = 2M x 4N, software-pipelined.
// ===========================================================================
__global__ void __launch_bounds__(256, 1)
k_combine_gemm(const bf16_t* __restrict__ colprobs,
               const bf16_t* __restrict__ yhat_t,
               float* __restrict__ y) {
  int w     = threadIdx.x >> 5;
  int mbase = (blockIdx.y * 8 + w) * 32;
  int nbase = blockIdx.x * 64;
  const bf16_t* pa0 = colprobs + (size_t)mbase * SLOTS_;
  const bf16_t* pa1 = pa0 + (size_t)16 * SLOTS_;
  const bf16_t* p0  = yhat_t + (size_t)(nbase +  0) * SLOTS_;
  const bf16_t* p1  = yhat_t + (size_t)(nbase + 16) * SLOTS_;
  const bf16_t* p2  = yhat_t + (size_t)(nbase + 32) * SLOTS_;
  const bf16_t* p3  = yhat_t + (size_t)(nbase + 48) * SLOTS_;

  v8f acc[8] = {};
  v16bf a0 = load_frag_row16_k32(pa0, SLOTS_);
  v16bf a1 = load_frag_row16_k32(pa1, SLOTS_);
  v16bf b0 = load_frag_col16_k32(p0, SLOTS_);
  v16bf b1 = load_frag_col16_k32(p1, SLOTS_);
  v16bf b2 = load_frag_col16_k32(p2, SLOTS_);
  v16bf b3 = load_frag_col16_k32(p3, SLOTS_);

  for (int k = 0; k < SLOTS_ - 32; k += 32) {
    int kn = k + 32;
    __builtin_prefetch(pa0 + kn + 32, 0, 3);
    v16bf a0n = load_frag_row16_k32(pa0 + kn, SLOTS_);
    v16bf a1n = load_frag_row16_k32(pa1 + kn, SLOTS_);
    v16bf b0n = load_frag_col16_k32(p0 + kn, SLOTS_);
    v16bf b1n = load_frag_col16_k32(p1 + kn, SLOTS_);
    v16bf b2n = load_frag_col16_k32(p2 + kn, SLOTS_);
    v16bf b3n = load_frag_col16_k32(p3 + kn, SLOTS_);
    acc[0] = wmma_bf16(a0, b0, acc[0]);
    acc[1] = wmma_bf16(a0, b1, acc[1]);
    acc[2] = wmma_bf16(a0, b2, acc[2]);
    acc[3] = wmma_bf16(a0, b3, acc[3]);
    acc[4] = wmma_bf16(a1, b0, acc[4]);
    acc[5] = wmma_bf16(a1, b1, acc[5]);
    acc[6] = wmma_bf16(a1, b2, acc[6]);
    acc[7] = wmma_bf16(a1, b3, acc[7]);
    a0 = a0n; a1 = a1n; b0 = b0n; b1 = b1n; b2 = b2n; b3 = b3n;
  }
  acc[0] = wmma_bf16(a0, b0, acc[0]);
  acc[1] = wmma_bf16(a0, b1, acc[1]);
  acc[2] = wmma_bf16(a0, b2, acc[2]);
  acc[3] = wmma_bf16(a0, b3, acc[3]);
  acc[4] = wmma_bf16(a1, b0, acc[4]);
  acc[5] = wmma_bf16(a1, b1, acc[5]);
  acc[6] = wmma_bf16(a1, b2, acc[6]);
  acc[7] = wmma_bf16(a1, b3, acc[7]);

  float* po  = y + (size_t)mbase * D_ + nbase;
  float* po2 = po + (size_t)16 * D_;
  store_tile_f32(po  +  0, D_, acc[0]);
  store_tile_f32(po  + 16, D_, acc[1]);
  store_tile_f32(po  + 32, D_, acc[2]);
  store_tile_f32(po  + 48, D_, acc[3]);
  store_tile_f32(po2 +  0, D_, acc[4]);
  store_tile_f32(po2 + 16, D_, acc[5]);
  store_tile_f32(po2 + 32, D_, acc[6]);
  store_tile_f32(po2 + 48, D_, acc[7]);
}

// ===========================================================================
// Host-side launch.  Inputs: tokens, expert_mixture, W1, b1, W2, b2, W3, b3.
// ===========================================================================
extern "C" void kernel_launch(void* const* d_in, const int* in_sizes, int n_in,
                              void* d_out, int out_size, void* d_ws, size_t ws_size,
                              hipStream_t stream) {
  const float* tokens = (const float*)d_in[0];   // [32768 x 1024]
  const float* em     = (const float*)d_in[1];   // [1024 x 512]
  const float* W1     = (const float*)d_in[2];   // [64 x 1024 x 256]
  const float* b1     = (const float*)d_in[3];   // [64 x 256]
  const float* W2     = (const float*)d_in[4];   // [64 x 256 x 256]
  const float* b2     = (const float*)d_in[5];   // [64 x 256]
  const float* W3     = (const float*)d_in[6];   // [64 x 256 x 1024]
  const float* b3     = (const float*)d_in[7];   // [64 x 1024]
  float* y_out        = (float*)d_out;           // [32768 x 1024]

  // ---- workspace partition ----
  char* ws = (char*)d_ws;
  bf16_t* tokens_bf = (bf16_t*)ws; ws += (size_t)BT_ * D_ * 2;            // [BT x D]
  bf16_t* tokens_t  = (bf16_t*)ws; ws += (size_t)D_ * BT_ * 2;            // [D x BT]
  bf16_t* em_t      = (bf16_t*)ws; ws += (size_t)SLOTS_ * D_ * 2;         // [S x D]
  bf16_t* w1t       = (bf16_t*)ws; ws += (size_t)NEXP_ * HID_ * D_ * 2;   // [e][H x D]
  bf16_t* w2t       = (bf16_t*)ws; ws += (size_t)NEXP_ * HID_ * HID_ * 2; // [e][H x H]
  bf16_t* w3t       = (bf16_t*)ws; ws += (size_t)NEXP_ * D_ * HID_ * 2;   // [e][D x H]
  float*  mat       = (float*)ws;  ws += (size_t)BT_ * SLOTS_ * 4;        // [BT x S]
  float*  rowmax    = (float*)ws;  ws += (size_t)BT_ * 4;
  float*  rowinv    = (float*)ws;  ws += (size_t)BT_ * 4;
  float*  colmax    = (float*)ws;  ws += (size_t)SLOTS_ * 4;
  float*  colinv    = (float*)ws;  ws += (size_t)SLOTS_ * 4;
  float*  pmax      = (float*)ws;  ws += (size_t)32 * SLOTS_ * 4;
  float*  psum      = (float*)ws;  ws += (size_t)32 * SLOTS_ * 4;
  bf16_t* colprobs  = (bf16_t*)ws; ws += (size_t)BT_ * SLOTS_ * 2;        // [BT x S]
  bf16_t* rowprobsT = (bf16_t*)ws; ws += (size_t)SLOTS_ * BT_ * 2;        // [S x BT]
  float*  xpart     = (float*)ws;  ws += (size_t)KSPLIT_ * SLOTS_ * D_ * 4;
  bf16_t* xhat      = (bf16_t*)ws; ws += (size_t)SLOTS_ * D_ * 2;         // [S x D]
  bf16_t* yhat_t    = (bf16_t*)ws; ws += (size_t)D_ * SLOTS_ * 2;         // [D x S]

  // 1) convert operands to bf16 (plain + K-contiguous transposed layouts)
  k_cvt_bf16<<<(BT_ * (size_t)D_) / (256 * 8), 256, 0, stream>>>(tokens, tokens_bf);
  k_transpose_cvt<<<dim3(D_ / 32, BT_ / 32, 1),        256, 0, stream>>>(tokens, tokens_t, BT_, D_);
  k_transpose_cvt<<<dim3(SLOTS_ / 32, D_ / 32, 1),     256, 0, stream>>>(em, em_t, D_, SLOTS_);
  k_transpose_cvt<<<dim3(HID_ / 32, D_ / 32, NEXP_),   256, 0, stream>>>(W1, w1t, D_, HID_);
  k_transpose_cvt<<<dim3(HID_ / 32, HID_ / 32, NEXP_), 256, 0, stream>>>(W2, w2t, HID_, HID_);
  k_transpose_cvt<<<dim3(D_ / 32, HID_ / 32, NEXP_),   256, 0, stream>>>(W3, w3t, HID_, D_);

  // 2) routing logits
  k_router_gemm<<<dim3(SLOTS_ / 64, BT_ / 256, 1), 256, 0, stream>>>(tokens_bf, em_t, mat);

  // 3-5) dual softmax + probability matrices
  k_row_stats<<<BT_ / 8, 256, 0, stream>>>(mat, rowmax, rowinv);
  k_col_partial<<<dim3(SLOTS_ / 32, BT_ / 1024, 1), 256, 0, stream>>>(mat, pmax, psum);
  k_col_final<<<SLOTS_ / 256, 256, 0, stream>>>(pmax, psum, colmax, colinv);
  k_probs<<<dim3(BT_ / 32, SLOTS_ / 32, 1), 256, 0, stream>>>(mat, rowmax, rowinv,
                                                              colmax, colinv,
                                                              colprobs, rowprobsT);

  // 6) dispatch: x_hat = row_probs^T @ tokens  (K-split x4 + reduce)
  k_dispatch_gemm<<<dim3(D_ / 64, SLOTS_ / 128, KSPLIT_), 256, 0, stream>>>(rowprobsT, tokens_t, xpart);
  k_xhat_reduce<<<(SLOTS_ * D_) / 256, 256, 0, stream>>>(xpart, xhat);

  // 7) per-expert MLP (one WGP per expert)
  k_expert_mlp<<<NEXP_, 256, 0, stream>>>(xhat, w1t, b1, w2t, b2, w3t, b3, yhat_t);

  // 8) combine: y = col_probs @ y_hat
  k_combine_gemm<<<dim3(D_ / 64, BT_ / 256, 1), 256, 0, stream>>>(colprobs, yhat_t, y_out);
}
